// Encoder_77043123356186
// MI455X (gfx1250) — matrix operand order repaired and checked by reading, not compiled, not run on hardware
//
#include <hip/hip_runtime.h>
#include <hip/hip_bf16.h>

typedef __attribute__((ext_vector_type(16))) __bf16 v16bf;
typedef __attribute__((ext_vector_type(8)))  float  v8f;

#define N_NODES 50000
#define IN_CH   256
#define HID     128
#define LAT     64

// ---------------------------------------------------------------------------
// Degree kernels
// ---------------------------------------------------------------------------
__global__ __launch_bounds__(256) void deg_init(float* __restrict__ deg, int n) {
    int i = blockIdx.x * 256 + threadIdx.x;
    if (i < n) deg[i] = 1.0f;               // self loop contributes 1 to every node
}

__global__ __launch_bounds__(256) void deg_accum(const int* __restrict__ dst,
                                                 float* __restrict__ deg,
                                                 int E, int n) {
    int e = blockIdx.x * 256 + threadIdx.x;
    if (e >= E) return;
    int d = dst[e];
    d = min(max(d, 0), n - 1);
    atomicAdd(&deg[d], 1.0f);
}

__global__ __launch_bounds__(256) void deg_to_rsqrt(float* __restrict__ deg, int n) {
    int i = blockIdx.x * 256 + threadIdx.x;
    if (i >= n) return;
    float d = deg[i];
    deg[i] = (d > 0.0f) ? rsqrtf(d) : 0.0f;
}

// ---------------------------------------------------------------------------
// bf16 WMMA GEMM:  H[M,N] = A[M,K] @ W[K,N]   (f32 in, bf16 MAC, f32 accum)
// 256 threads = 8 waves; each wave computes one 16-row M tile across all N.
// W is staged to LDS pre-swizzled into WMMA B-fragment order:
//   wlds viewed as v16bf[K/32][N/16][32]: entry (kc, nt, lane) is the exact
//   16-element contiguous fragment lane `lane` feeds to the WMMA — so the
//   inner loop is two ds_load_b128 per fragment, no per-element packing.
// B-fragment layout (16-bit, 32x16): lane l covers column n = nt*16 + (l&15),
//   K rows kc*32 + (j>>3)*16 + (l>>4)*8 + (j&7), j = 0..15.
// ---------------------------------------------------------------------------
template <int K, int N>
__global__ __launch_bounds__(256) void gemm_bf16_wmma(const float* __restrict__ A,
                                                      const float* __restrict__ W,
                                                      float* __restrict__ H,
                                                      int M) {
    constexpr int NT = N / 16;
    constexpr int KC = K / 32;
    __shared__ alignas(32) __bf16 wlds[K * N];

    // Swizzle W (row-major f32 [K][N]) into B-fragment order (bf16).
    for (int i = threadIdx.x; i < K * N; i += 256) {
        const int j    = i & 15;            // element within lane fragment
        const int lane = (i >> 4) & 31;     // lane
        const int nt   = (i >> 9) % NT;     // N tile
        const int kc   = i / (512 * NT);    // K chunk of 32
        const int n = nt * 16 + (lane & 15);
        const int k = kc * 32 + (j >> 3) * 16 + (lane >> 4) * 8 + (j & 7);
        wlds[i] = (__bf16)W[k * N + n];
    }
    __syncthreads();

    const int wave = threadIdx.x >> 5;
    const int lane = threadIdx.x & 31;
    const int m0   = (blockIdx.x * 8 + wave) * 16;
    if (m0 >= M) return;                    // wave-uniform: EXEC stays all-1s

    const int half = lane >> 4;             // 0: lanes 0-15, 1: lanes 16-31
    const int l16  = lane & 15;

    v8f acc[NT] = {};
    const v16bf* __restrict__ wfrag = reinterpret_cast<const v16bf*>(wlds);
    const float* arow = A + (size_t)(m0 + l16) * K;

    for (int kc = 0; kc < KC; ++kc) {
        // A fragment per documented 16-bit 16x32 layout:
        //  slots 0..7  = K = kc*32 + half*8 + 0..7
        //  slots 8..15 = K = kc*32 + 16 + half*8 + 0..7
        const int kb = kc * 32 + half * 8;
        float4 c0 = *(const float4*)(arow + kb + 0);
        float4 c1 = *(const float4*)(arow + kb + 4);
        float4 c2 = *(const float4*)(arow + kb + 16);
        float4 c3 = *(const float4*)(arow + kb + 20);
        v16bf af;
        af[0]  = (__bf16)c0.x; af[1]  = (__bf16)c0.y; af[2]  = (__bf16)c0.z; af[3]  = (__bf16)c0.w;
        af[4]  = (__bf16)c1.x; af[5]  = (__bf16)c1.y; af[6]  = (__bf16)c1.z; af[7]  = (__bf16)c1.w;
        af[8]  = (__bf16)c2.x; af[9]  = (__bf16)c2.y; af[10] = (__bf16)c2.z; af[11] = (__bf16)c2.w;
        af[12] = (__bf16)c3.x; af[13] = (__bf16)c3.y; af[14] = (__bf16)c3.z; af[15] = (__bf16)c3.w;

        #pragma unroll
        for (int nt = 0; nt < NT; ++nt) {
            const v16bf bf = wfrag[(kc * NT + nt) * 32 + lane];  // 2x ds_load_b128
            acc[nt] = __builtin_amdgcn_wmma_f32_16x16x32_bf16(
                false, af, false, bf, (short)0, acc[nt], false, false);
        }
    }

    // C/D layout: VGPR r, lanes 0-15 -> M = r, lanes 16-31 -> M = 8 + r
    #pragma unroll
    for (int nt = 0; nt < NT; ++nt) {
        const int n = nt * 16 + l16;
        #pragma unroll
        for (int r = 0; r < 8; ++r)
            H[(size_t)(m0 + half * 8 + r) * N + n] = acc[nt][r];
    }
}

// ---------------------------------------------------------------------------
// Aggregation:  out[d] += dis[s]*dis[d] * h[s]  over edges; self loops via init
// ---------------------------------------------------------------------------
template <int F>
__global__ __launch_bounds__(256) void agg_selfloop_init(const float* __restrict__ h,
                                                         const float* __restrict__ dis,
                                                         float* __restrict__ out,
                                                         int n) {
    int i = blockIdx.x * 256 + threadIdx.x;
    if (i >= n * F) return;
    int node = i / F;
    float w = dis[node] * dis[node];
    out[i] = w * h[i];
}

template <int F>
__global__ __launch_bounds__(256) void agg_edges(const float* __restrict__ h,
                                                 const int* __restrict__ srcIdx,
                                                 const int* __restrict__ dstIdx,
                                                 const float* __restrict__ dis,
                                                 float* __restrict__ out,
                                                 int E, int n) {
    constexpr int TPE = F / 4;              // threads per edge, float4 each
    int idx = blockIdx.x * 256 + threadIdx.x;
    int e   = idx / TPE;
    int fo  = (idx % TPE) * 4;
    if (e >= E) return;
    int s = srcIdx[e];
    int d = dstIdx[e];
    s = min(max(s, 0), n - 1);
    d = min(max(d, 0), n - 1);
    float w = dis[s] * dis[d];
    float4 v = *(const float4*)(h + (size_t)s * F + fo);
    float* o = out + (size_t)d * F + fo;
    atomicAdd(o + 0, w * v.x);
    atomicAdd(o + 1, w * v.y);
    atomicAdd(o + 2, w * v.z);
    atomicAdd(o + 3, w * v.w);
}

template <int F>
__global__ __launch_bounds__(256) void bias_relu(float* __restrict__ a,
                                                 const float* __restrict__ b,
                                                 int n) {
    int i = blockIdx.x * 256 + threadIdx.x;
    if (i >= n * F) return;
    float v = a[i] + b[i % F];
    a[i] = v > 0.0f ? v : 0.0f;
}

// ---------------------------------------------------------------------------
// Launcher
// ---------------------------------------------------------------------------
extern "C" void kernel_launch(void* const* d_in, const int* in_sizes, int n_in,
                              void* d_out, int out_size, void* d_ws, size_t ws_size,
                              hipStream_t stream) {
    const float* x   = (const float*)d_in[0];
    const int*   ei  = (const int*)d_in[1];   // [2, E] flat: src = ei[0..E), dst = ei[E..2E)
    const float* W1  = (const float*)d_in[2];
    const float* b1  = (const float*)d_in[3];
    const float* W2  = (const float*)d_in[4];
    const float* b2  = (const float*)d_in[5];
    float*       out = (float*)d_out;

    const int n = in_sizes[0] / IN_CH;        // 50000
    const int E = in_sizes[1] / 2;            // 800000
    const int* src = ei;
    const int* dst = ei + E;

    // workspace layout (bytes, 256-aligned)
    char* ws = (char*)d_ws;
    size_t off = 0;
    float* deg = (float*)(ws + off); off += ((size_t)n * 4 + 255) & ~(size_t)255;       // deg -> dis
    float* h1  = (float*)(ws + off); off += ((size_t)n * HID * 4 + 255) & ~(size_t)255; // x @ W1
    float* g1  = (float*)(ws + off); off += ((size_t)n * HID * 4 + 255) & ~(size_t)255; // agg1 -> a1
    float* h2  = (float*)(ws + off); off += ((size_t)n * LAT * 4 + 255) & ~(size_t)255; // a1 @ W2
    (void)ws_size; (void)n_in; (void)out_size;

    const int B = 256;
    auto cdiv = [](long long a, long long b) { return (int)((a + b - 1) / b); };

    // degrees -> deg^{-1/2}
    deg_init<<<cdiv(n, B), B, 0, stream>>>(deg, n);
    deg_accum<<<cdiv(E, B), B, 0, stream>>>(dst, deg, E, n);
    deg_to_rsqrt<<<cdiv(n, B), B, 0, stream>>>(deg, n);

    // ---- layer 1 ----
    const int mtiles = cdiv(n, 16);
    gemm_bf16_wmma<IN_CH, HID><<<cdiv(mtiles, 8), B, 0, stream>>>(x, W1, h1, n);
    agg_selfloop_init<HID><<<cdiv((long long)n * HID, B), B, 0, stream>>>(h1, deg, g1, n);
    agg_edges<HID><<<cdiv((long long)E * (HID / 4), B), B, 0, stream>>>(h1, src, dst, deg, g1, E, n);
    bias_relu<HID><<<cdiv((long long)n * HID, B), B, 0, stream>>>(g1, b1, n);

    // ---- layer 2 ----
    gemm_bf16_wmma<HID, LAT><<<cdiv(mtiles, 8), B, 0, stream>>>(g1, W2, h2, n);
    agg_selfloop_init<LAT><<<cdiv((long long)n * LAT, B), B, 0, stream>>>(h2, deg, out, n);
    agg_edges<LAT><<<cdiv((long long)E * (LAT / 4), B), B, 0, stream>>>(h2, src, dst, deg, out, E, n);
    bias_relu<LAT><<<cdiv((long long)n * LAT, B), B, 0, stream>>>(out, b2, n);
}
